// quadjetResNetBlock_58763742544551
// MI455X (gfx1250) — matrix operand order repaired and checked by reading, not compiled, not run on hardware
//
#include <hip/hip_runtime.h>
#include <hip/hip_bf16.h>

// ---------------------------------------------------------------------------
// quadjet ResNet block, fused 3-layer bf16-WMMA kernel for gfx1250 (MI455X)
//
// Layer GEMM:  out[m,o] = sum_k A[m,k] * B[k,o] + b[o],   m=(n,g), K=768
//   k in [0,512):  A = x[n, k/2, 2g + k%2]      (constant across layers)
//   k in [512,768):A = q_prev[n, k-512, g]      (changes per layer)
// Weights prepacked bf16 as B[layer][kb][o][kk] (kb: 24 blocks of K=32).
// Weight K-blocks are DMA'd global->LDS with GLOBAL_LOAD_ASYNC_TO_LDS_B128
// (ASYNCcnt), double-buffered one block ahead, pipelined across layers.
// ---------------------------------------------------------------------------

typedef __attribute__((ext_vector_type(16))) __bf16 v16bf;
typedef __attribute__((ext_vector_type(8)))  __bf16 v8bf;
typedef __attribute__((ext_vector_type(8)))  float  v8f;

#define TN   32            // n-rows per workgroup
#define MT   96            // M rows per workgroup (TN*3)
#define XSS  520           // xs row stride in halfwords (512 + 8 pad)
#define QSS  264           // qs row stride in halfwords (256 + 8 pad)
#define BSS  40            // bs row stride in halfwords (32  + 8 pad)
#define XS_BYTES (MT * XSS * 2)      //  99840 (also reused as f32 out-stage)
#define QS_BYTES (MT * QSS * 2)      //  50688
#define Q0_BYTES (TN * 256 * 3 * 4)  //  98304 residual (f32, global layout)
#define BS_BYTES (256 * BSS * 2)     //  20480 per buffer
#define QS_OFF   (XS_BYTES)                        //  99840
#define Q0_OFF   (XS_BYTES + QS_BYTES)             // 150528
#define BS_OFF   (Q0_OFF + Q0_BYTES)               // 248832
#define SMEM_BYTES (BS_OFF + 2 * BS_BYTES)         // 289792 < 320KB/WGP
#define KB_TOTAL 24        // 768 / 32
#define WPACK    8192      // halfwords per K-block of packed B (256*32)

__device__ __forceinline__ __bf16 f2bf(float f) {
  union { float f; unsigned u; } a; a.f = f;
  unsigned u = a.u + 0x7FFFu + ((a.u >> 16) & 1u);   // round-to-nearest-even
  unsigned short h = (unsigned short)(u >> 16);
  __bf16 r; __builtin_memcpy(&r, &h, 2); return r;
}

// CDNA5 async DMA: global -> LDS, 16B per lane, tracked by ASYNCcnt.
__device__ __forceinline__ void async_g2l_b128(void* lds_ptr, const void* gptr) {
  unsigned lds_off = (unsigned)(size_t)lds_ptr;     // low 32b = LDS byte addr
  unsigned long long ga = (unsigned long long)(size_t)gptr;
  asm volatile("global_load_async_to_lds_b128 %0, %1, off"
               :: "v"(lds_off), "v"(ga) : "memory");
}
#define ASYNC_WAIT_4() asm volatile("s_wait_asynccnt 0x4" ::: "memory")
#define ASYNC_WAIT_0() asm volatile("s_wait_asynccnt 0x0" ::: "memory")

// ---------------------------------------------------------------------------
// Prep: pack W1/W2/W3 (f32 [256,256,3]) -> bf16 B[layer][kb][o][kk]
// ---------------------------------------------------------------------------
__global__ void prep_weights(const float* __restrict__ W1,
                             const float* __restrict__ W2,
                             const float* __restrict__ W3,
                             __bf16* __restrict__ Bp) {
  int idx = blockIdx.x * 256 + threadIdx.x;
  if (idx >= 3 * KB_TOTAL * WPACK) return;
  int layer = idx / (KB_TOTAL * WPACK);
  int rem   = idx % (KB_TOTAL * WPACK);
  int kb    = rem / WPACK;
  int r2    = rem % WPACK;
  int o     = r2 >> 5;
  int kk    = r2 & 31;
  int k     = kb * 32 + kk;
  const float* W = (layer == 0) ? W1 : (layer == 1) ? W2 : W3;
  float v;
  if (k < 512) { int c = k >> 1, j = k & 1; v = W[(o * 256 + c) * 3 + j]; }
  else         { int c = k - 512;           v = W[(o * 256 + c) * 3 + 2]; }
  Bp[idx] = f2bf(v);
}

// ---------------------------------------------------------------------------
// Fused 3-layer block. 256 threads (8 waves), 2048 workgroups.
// ---------------------------------------------------------------------------
__global__ __launch_bounds__(256, 1)
void quadjet_fused_kernel(const float* __restrict__ x,
                          const float* __restrict__ q,
                          const __bf16* __restrict__ Bp,
                          const float* __restrict__ b1,
                          const float* __restrict__ b2,
                          const float* __restrict__ b3,
                          float* __restrict__ out) {
  extern __shared__ __align__(16) char smem[];
  __bf16* xs = (__bf16*)smem;                 // [96][XSS]  A, K<512 (bf16)
  __bf16* qs = (__bf16*)(smem + QS_OFF);      // [96][QSS]  A, K>=512 (bf16)
  float*  q0 = (float*)(smem + Q0_OFF);       // residual, global f32 layout
  // bs double buffer at BS_OFF: [2][256][BSS] bf16

  const int tid  = threadIdx.x;
  const int lane = tid & 31;
  const int wave = tid >> 5;
  const int l15  = lane & 15;
  const int hi   = (lane >> 4) & 1;     // lane half selects K sub-range
  const int n0   = blockIdx.x * TN;

  // issue async DMA of one weight K-block into LDS buffer `buf`
  auto issue_block = [&](int layer, int kb, int buf) {
    const __bf16* src = Bp + ((size_t)layer * KB_TOTAL + kb) * WPACK;
    char* dst = smem + BS_OFF + buf * BS_BYTES;
#pragma unroll
    for (int i = 0; i < 4; ++i) {
      int chunk = tid * 4 + i;          // 1024 x 16B chunks
      int o  = chunk >> 2;
      int kq = (chunk & 3) * 8;
      async_g2l_b128(dst + (o * BSS + kq) * 2, src + chunk * 8);
    }
  };

  // kick off layer-0 block-0 DMA before doing anything else
  issue_block(0, 0, 0);

  // ---- Phase 0: stage x (bf16), q (bf16) and residual q (f32) into LDS ----
  for (int p = tid; p < TN * 256; p += 256) {
    int ln = p >> 8, c = p & 255;
    const float* xsrc = x + ((size_t)(n0 + ln) * 256 + c) * 6;
    float2 v01 = *(const float2*)(xsrc + 0);
    float2 v23 = *(const float2*)(xsrc + 2);
    float2 v45 = *(const float2*)(xsrc + 4);
    __bf16* r0 = xs + (ln * 3 + 0) * XSS + c * 2;
    __bf16* r1 = xs + (ln * 3 + 1) * XSS + c * 2;
    __bf16* r2 = xs + (ln * 3 + 2) * XSS + c * 2;
    r0[0] = f2bf(v01.x); r0[1] = f2bf(v01.y);
    r1[0] = f2bf(v23.x); r1[1] = f2bf(v23.y);
    r2[0] = f2bf(v45.x); r2[1] = f2bf(v45.y);
    const float* qsrc = q + ((size_t)(n0 + ln) * 256 + c) * 3;
    float q0v = qsrc[0], q1v = qsrc[1], q2v = qsrc[2];
    qs[(ln * 3 + 0) * QSS + c] = f2bf(q0v);
    qs[(ln * 3 + 1) * QSS + c] = f2bf(q1v);
    qs[(ln * 3 + 2) * QSS + c] = f2bf(q2v);
    float* qr = q0 + (ln * 256 + c) * 3;
    qr[0] = q0v; qr[1] = q1v; qr[2] = q2v;
  }
  __syncthreads();

  union BF { v16bf v; struct { v8bf lo, hi; } s; };
  const v8f vzero = {0.f, 0.f, 0.f, 0.f, 0.f, 0.f, 0.f, 0.f};

  for (int layer = 0; layer < 3; ++layer) {
    v8f acc[6][2];
#pragma unroll
    for (int mt = 0; mt < 6; ++mt) { acc[mt][0] = vzero; acc[mt][1] = vzero; }

    for (int kb = 0; kb < KB_TOTAL; ++kb) {
      // ---- keep the DMA pipeline one K-block ahead (across layers too) ----
      if (kb + 1 < KB_TOTAL) {
        issue_block(layer, kb + 1, (kb + 1) & 1);
        ASYNC_WAIT_4();                 // current block done, next in flight
      } else if (layer < 2) {
        issue_block(layer + 1, 0, 0);   // parity continues: 24 & 1 == 0
        ASYNC_WAIT_4();
      } else {
        ASYNC_WAIT_0();
      }
      __syncthreads();

      const __bf16* bs = (const __bf16*)(smem + BS_OFF + (kb & 1) * BS_BYTES);

      // ---- A source: x-part for kb<16, q-part after ----
      const __bf16* abase; int astr, acol;
      if (kb < 16) { abase = xs; astr = XSS; acol = kb * 32; }
      else         { abase = qs; astr = QSS; acol = (kb - 16) * 32; }

      // ---- B fragments: lane holds 16 contiguous K for column l15 ----
      BF bf0, bf1;
      {
        const __bf16* br0 = bs + ((wave * 2 + 0) * 16 + l15) * BSS + hi * 16;
        const __bf16* br1 = bs + ((wave * 2 + 1) * 16 + l15) * BSS + hi * 16;
        bf0.s.lo = *(const v8bf*)(br0);  bf0.s.hi = *(const v8bf*)(br0 + 8);
        bf1.s.lo = *(const v8bf*)(br1);  bf1.s.hi = *(const v8bf*)(br1 + 8);
      }

      // ---- 6 M-tiles, A frags double-buffered to hide ds latency ----
      const __bf16* ar0 = abase + l15 * astr + acol + hi * 16;
      BF af[2];
      af[0].s.lo = *(const v8bf*)(ar0);
      af[0].s.hi = *(const v8bf*)(ar0 + 8);
#pragma unroll
      for (int mt = 0; mt < 6; ++mt) {
        if (mt + 1 < 6) {
          const __bf16* ar = abase + ((mt + 1) * 16 + l15) * astr + acol + hi * 16;
          af[(mt + 1) & 1].s.lo = *(const v8bf*)(ar);
          af[(mt + 1) & 1].s.hi = *(const v8bf*)(ar + 8);
        }
        acc[mt][0] = __builtin_amdgcn_wmma_f32_16x16x32_bf16(
            false, af[mt & 1].v, false, bf0.v, (short)0, acc[mt][0], false, false);
        acc[mt][1] = __builtin_amdgcn_wmma_f32_16x16x32_bf16(
            false, af[mt & 1].v, false, bf1.v, (short)0, acc[mt][1], false, false);
      }
      __syncthreads();
    }

    // ---- epilogue ----
    const float* bvec = (layer == 0) ? b1 : (layer == 1) ? b2 : b3;
    float bias[2];
    bias[0] = bvec[(wave * 2 + 0) * 16 + l15];
    bias[1] = bvec[(wave * 2 + 1) * 16 + l15];

    if (layer < 2) {
      // ReLU, write back as bf16 into qs (A q-part for the next layer)
#pragma unroll
      for (int mt = 0; mt < 6; ++mt)
#pragma unroll
        for (int nt = 0; nt < 2; ++nt) {
          int o = (wave * 2 + nt) * 16 + l15;
#pragma unroll
          for (int r = 0; r < 8; ++r) {
            int m = mt * 16 + r + hi * 8;   // C layout: VGPR r -> M=r / r+8
            float v = acc[mt][nt][r] + bias[nt];
            qs[m * QSS + o] = f2bf(v > 0.f ? v : 0.f);
          }
        }
      __syncthreads();
    } else {
      // stage conv3+bias into xs region (dead now) in global f32 layout,
      // then one coalesced LDS->global sweep with fused residual + ReLU.
      float* stage = (float*)smem;          // 96*256 f32 = 98304B <= XS_BYTES
#pragma unroll
      for (int mt = 0; mt < 6; ++mt)
#pragma unroll
        for (int nt = 0; nt < 2; ++nt) {
          int o = (wave * 2 + nt) * 16 + l15;
#pragma unroll
          for (int r = 0; r < 8; ++r) {
            int m  = mt * 16 + r + hi * 8;
            int ln = m / 3, g = m % 3;
            stage[(ln * 256 + o) * 3 + g] = acc[mt][nt][r] + bias[nt];
          }
        }
      __syncthreads();

      const float4* sv = (const float4*)stage;
      const float4* rv = (const float4*)q0;
      float4* ov = (float4*)(out + (size_t)n0 * 768);
      for (int i = tid; i < TN * 256 * 3 / 4; i += 256) {   // 6144 chunks
        float4 a = sv[i], b = rv[i], r;
        r.x = a.x + b.x; r.y = a.y + b.y; r.z = a.z + b.z; r.w = a.w + b.w;
        r.x = r.x > 0.f ? r.x : 0.f;
        r.y = r.y > 0.f ? r.y : 0.f;
        r.z = r.z > 0.f ? r.z : 0.f;
        r.w = r.w > 0.f ? r.w : 0.f;
        ov[i] = r;
      }
    }
  }
}

// ---------------------------------------------------------------------------
extern "C" void kernel_launch(void* const* d_in, const int* in_sizes, int n_in,
                              void* d_out, int out_size, void* d_ws, size_t ws_size,
                              hipStream_t stream) {
  const float* x  = (const float*)d_in[0];
  const float* q  = (const float*)d_in[1];
  const float* W1 = (const float*)d_in[2];
  const float* b1 = (const float*)d_in[3];
  const float* W2 = (const float*)d_in[4];
  const float* b2 = (const float*)d_in[5];
  const float* W3 = (const float*)d_in[6];
  const float* b3 = (const float*)d_in[7];
  float* out = (float*)d_out;

  __bf16* Bp = (__bf16*)d_ws;   // 3*24*8192 bf16 = 1.18 MB packed weights

  int wtot = 3 * KB_TOTAL * WPACK;
  prep_weights<<<(wtot + 255) / 256, 256, 0, stream>>>(W1, W2, W3, Bp);

  (void)hipFuncSetAttribute((const void*)quadjet_fused_kernel,
                            hipFuncAttributeMaxDynamicSharedMemorySize,
                            (int)SMEM_BYTES);
  const int nblocks = 65536 / TN;   // 2048
  quadjet_fused_kernel<<<nblocks, 256, SMEM_BYTES, stream>>>(
      x, q, Bp, b1, b2, b3, out);
}
